// Rank_Model_75591424410215
// MI455X (gfx1250) — compile-verified, matrix operand order checked by reading, not run on hardware
//
#include <hip/hip_runtime.h>
#include <hip/hip_bf16.h>
#include <math.h>

typedef __attribute__((ext_vector_type(16))) _Float16 v16h;
typedef __attribute__((ext_vector_type(8)))  float    v8f;

#define NSF   16
#define NBATCH 16
#define NLI   128
#define NTG   1024
#define NLIG  2048
#define NREC  16384
#define ELIG  32768
#define EREC  196608
#define NGAUSS 32
#define NPAIR (NBATCH*NLI*NTG)   /* 2097152 */

// ---------------- WMMA helper ----------------

__device__ inline v8f wmma_f16(v16h a, v16h b, v8f c) {
  return __builtin_amdgcn_wmma_f32_16x16x32_f16(false, a, false, b, (short)0, c,
                                                false, false);
}

// gfx1250 16-bit operand lane layout: lane%16 = row(A)/col(B); wave-half selects
// K subset; K(e) = kt*32 + e + 8*(e>=8) + 8*half.
// We keep all staged operands in "fragment order": frag[(tile)*32 + lane][e]
// so a fragment load is one contiguous 32B read per lane.
__device__ inline int frag_pos_of(int k /*0..31 within kt*/, int row) {
  int hf = (k >> 3) & 1;
  int ee = (k & 7) + ((k >= 16) ? 8 : 0);
  return (row + 16 * hf) * 16 + ee;   // within one 512-half fragment tile
}

// ---------------- small utility kernels ----------------

__global__ void fill_f32(float* p, long n) {
  long i = (long)blockIdx.x * blockDim.x + threadIdx.x;
  if (i < n) p[i] = 0.f;
}

// Pack a row-major f32 [K,N] weight into B-fragment order f16 panels:
// dst[((nt*KT + kt)*32 + lane)*16 + e], zero-padded beyond K,N.
__global__ void pack_b_frags(const float* src, int K, int N, int KT, int NT,
                             _Float16* dst) {
  int i = blockIdx.x * blockDim.x + threadIdx.x;
  int total = NT * KT * 512;
  if (i >= total) return;
  int e = i & 15;
  int lane = (i >> 4) & 31;
  int rest = i >> 9;
  int kt = rest % KT;
  int nt = rest / KT;
  int col = nt * 16 + (lane & 15);
  int half = lane >> 4;
  int k = kt * 32 + e + ((e >= 8) ? 8 : 0) + half * 8;
  float v = (k < K && col < N) ? src[k * N + col] : 0.f;
  dst[i] = (_Float16)v;
}

// concat the three head matrices [16,10] each into f32 [16,32] (cols 30,31 zero)
__global__ void build_whead(const float* piW, const float* sgW, const float* muW,
                            float* dst) {
  int i = blockIdx.x * blockDim.x + threadIdx.x;
  if (i >= 512) return;
  int r = i >> 5, c = i & 31;
  float v = 0.f;
  if (c < 10)      v = piW[r * 10 + c];
  else if (c < 20) v = sgW[r * 10 + c - 10];
  else if (c < 30) v = muW[r * 10 + c - 20];
  dst[i] = v;
}

// node buffer init: [N][48], first `dim` cols from src, rest zero
__global__ void node_init(const float* src, float* dst, int N, int dim) {
  int i = blockIdx.x * blockDim.x + threadIdx.x;
  if (i >= N * 48) return;
  int n = i / 48, c = i % 48;
  dst[i] = (c < dim) ? src[n * dim + c] : 0.f;
}
__global__ void node_update(const float* node_old, const float* agg,
                            const float* cnt, float* node_new, int N) {
  int i = blockIdx.x * blockDim.x + threadIdx.x;
  if (i >= N * 48) return;
  int n = i / 48;
  node_new[i] = node_old[i] + agg[i] / fmaxf(cnt[n], 1.f);
}

// ---------------- edge features: smear + MLP + spherical harmonics ----------------

__global__ void edge_feat_kernel(const float* pos, const int* ei, const float* bond,
                                 int nbond, float stop,
                                 const float* W1, const float* b1,
                                 const float* W2, const float* b2,
                                 float* ea_out, float* sh_out, int E) {
  int e = blockIdx.x * blockDim.x + threadIdx.x;
  if (e >= E) return;
  int s = ei[e], d = ei[E + e];
  float vx = pos[d * 3 + 0] - pos[s * 3 + 0];
  float vy = pos[d * 3 + 1] - pos[s * 3 + 1];
  float vz = pos[d * 3 + 2] - pos[s * 3 + 2];
  float r = sqrtf(vx * vx + vy * vy + vz * vz);
  float inv = 1.f / fmaxf(r, 1e-12f);
  float x = vx * inv, y = vy * inv, z = vz * inv;
  const float s3 = 1.7320508f, c15 = 3.8729833f, s5h = 1.1180340f;
  sh_out[e * 9 + 0] = 1.f;
  sh_out[e * 9 + 1] = s3 * x;
  sh_out[e * 9 + 2] = s3 * y;
  sh_out[e * 9 + 3] = s3 * z;
  sh_out[e * 9 + 4] = c15 * x * y;
  sh_out[e * 9 + 5] = c15 * y * z;
  sh_out[e * 9 + 6] = s5h * (3.f * z * z - 1.f);
  sh_out[e * 9 + 7] = c15 * x * z;
  sh_out[e * 9 + 8] = 0.5f * c15 * (x * x - y * y);

  float f[NGAUSS + 4];
  int nf = 0;
  for (int c = 0; c < nbond; ++c) f[nf++] = bond[e * nbond + c];
  float step = stop / (NGAUSS - 1);
  float coeff = -0.5f / (step * step);
  for (int g = 0; g < NGAUSS; ++g) {
    float dd = r - step * g;
    f[nf++] = expf(coeff * dd * dd);
  }
  float h[NSF];
  for (int c = 0; c < NSF; ++c) {
    float acc = b1[c];
    for (int k = 0; k < nf; ++k) acc += f[k] * W1[k * NSF + c];
    h[c] = fmaxf(acc, 0.f);
  }
  for (int c = 0; c < NSF; ++c) {
    float acc = b2[c];
    for (int k = 0; k < NSF; ++k) acc += h[k] * W2[k * NSF + c];
    ea_out[e * NSF + c] = acc;
  }
}

// ---------------- per-edge tensor product + scatter ----------------

__constant__ float C112c[3][3][5] = {
  { {0.f,0.f,-0.40824829f,0.f,-0.70710678f}, {0.f,0.70710678f,0.f,0.f,0.f}, {0.70710678f,0.f,0.f,0.f,0.f} },
  { {0.f,0.70710678f,0.f,0.f,0.f}, {0.f,0.f,0.81649658f,0.f,0.f}, {0.f,0.f,0.f,0.70710678f,0.f} },
  { {0.70710678f,0.f,0.f,0.f,0.f}, {0.f,0.f,0.f,0.70710678f,0.f}, {0.f,0.f,-0.40824829f,0.f,0.70710678f} }
};

__device__ void tp_scatter(int layer, const float* node, int srcIdx, int dstIdx,
                           const float* sh, const float* w, float* agg, float* cnt) {
  float x0[16];
#pragma unroll
  for (int u = 0; u < 16; ++u) x0[u] = node[dstIdx * 48 + u];
  float out[40];
#pragma unroll
  for (int c = 0; c < 40; ++c) out[c] = 0.f;

  if (layer == 0) {
    const float pw0 = 0.25f, pw1 = 0.25f;
    for (int wv = 0; wv < 16; ++wv) {
      float acc = 0.f;
      for (int u = 0; u < 16; ++u) acc += x0[u] * w[u * 16 + wv];
      out[wv] += pw0 * acc * sh[0];
    }
    for (int wv = 0; wv < 4; ++wv) {
      float acc = 0.f;
      for (int u = 0; u < 16; ++u) acc += x0[u] * w[256 + u * 4 + wv];
      for (int k = 0; k < 3; ++k) out[16 + wv * 3 + k] += pw1 * acc * sh[1 + k];
    }
    for (int c = 0; c < 28; ++c) atomicAdd(&agg[srcIdx * 48 + c], out[c]);
  } else {
    float x1[4][3];
    for (int u = 0; u < 4; ++u)
      for (int i = 0; i < 3; ++i) x1[u][i] = node[dstIdx * 48 + 16 + u * 3 + i];
    const float pw0 = 0.22360680f, pw1 = 0.20412415f, pw2 = 0.5f;
    const float c110 = 0.57735027f;
    for (int wv = 0; wv < 16; ++wv) {
      float acc = 0.f;
      for (int u = 0; u < 16; ++u) acc += x0[u] * w[u * 16 + wv];
      out[wv] += pw0 * acc * sh[0];
    }
    for (int wv = 0; wv < 4; ++wv) {
      float acc = 0.f;
      for (int u = 0; u < 16; ++u) acc += x0[u] * w[256 + u * 4 + wv];
      for (int k = 0; k < 3; ++k) out[16 + wv * 3 + k] += pw1 * acc * sh[1 + k];
    }
    for (int wv = 0; wv < 4; ++wv)
      for (int i = 0; i < 3; ++i) {
        float acc = 0.f;
        for (int u = 0; u < 4; ++u) acc += x1[u][i] * w[320 + u * 4 + wv];
        out[16 + wv * 3 + i] += pw1 * acc * sh[0];
      }
    for (int wv = 0; wv < 16; ++wv) {
      float acc = 0.f;
      for (int i = 0; i < 3; ++i) {
        float xw = 0.f;
        for (int u = 0; u < 4; ++u) xw += x1[u][i] * w[336 + u * 16 + wv];
        acc += xw * sh[1 + i];
      }
      out[wv] += pw0 * c110 * acc;
    }
    for (int wv = 0; wv < 4; ++wv) {
      float xw[3];
      for (int i = 0; i < 3; ++i) {
        xw[i] = 0.f;
        for (int u = 0; u < 4; ++u) xw[i] += x1[u][i] * w[400 + u * 4 + wv];
      }
      for (int k = 0; k < 3; ++k) {
        float acc = 0.f;
        for (int i = 0; i < 3; ++i)
          for (int j = 0; j < 3; ++j) acc += xw[i] * sh[1 + j] * C112c[i][j][k];
        out[28 + wv * 3 + k] += pw2 * acc;
      }
    }
    for (int wv = 0; wv < 4; ++wv) {
      float xw[3];
      for (int i = 0; i < 3; ++i) {
        xw[i] = 0.f;
        for (int u = 0; u < 4; ++u) xw[i] += x1[u][i] * w[416 + u * 4 + wv];
      }
      for (int k = 0; k < 3; ++k) {
        float acc = 0.f;
        for (int i = 0; i < 3; ++i)
          for (int j = 0; j < 5; ++j) acc += xw[i] * sh[4 + j] * C112c[i][k][j];
        out[16 + wv * 3 + k] += pw1 * acc;
      }
    }
    for (int c = 0; c < 40; ++c) atomicAdd(&agg[srcIdx * 48 + c], out[c]);
  }
  atomicAdd(&cnt[srcIdx], 1.f);
}

// ---------------- fused conv: edge-weight MLP (WMMA) + tensor product scatter ----------------
// One wave per 16-edge tile; all f16 operands staged in fragment order.

__global__ void __launch_bounds__(32)
conv_fused(const float* node, const int* ei, int E,
           const float* ea, const float* sh,
           const _Float16* W1p, const float* bias1,
           const _Float16* W2p, const float* bias2,
           int wn, int layer, float* agg, float* cnt) {
  __shared__ __align__(32) _Float16 eF[2 * 512];   // e_ tile, 2 K-tiles, frag order
  __shared__ __align__(32) _Float16 hF[2 * 512];   // hidden tile, frag order
  __shared__ float wL[16 * 432];                   // TP weights, row-major
  int tile = blockIdx.x;
  int lane = threadIdx.x;
  int n = lane & 15, half = lane >> 4;

  for (int idx = lane; idx < 1024; idx += 32) hF[idx] = (_Float16)0.f;

  for (int idx = lane; idx < 16 * 64; idx += 32) {
    int r = idx >> 6, c = idx & 63;
    int e = tile * 16 + r;
    float v = 0.f;
    if (e < E) {
      if (c < 16)      v = ea[e * 16 + c];
      else if (c < 32) v = node[ei[e] * 48 + (c - 16)];
      else if (c < 48) v = node[ei[E + e] * 48 + (c - 32)];
    }
    int kt = c >> 5;
    eF[kt * 512 + frag_pos_of(c & 31, r)] = (_Float16)v;
  }
  __syncthreads();

  const v16h* eFv = (const v16h*)eF;
  v16h a0 = eFv[lane];
  v16h a1 = eFv[32 + lane];
  const v16h* W1f = (const v16h*)W1p;   // KT=2, NT=3
#pragma unroll
  for (int nt = 0; nt < 3; ++nt) {
    v16h bf0 = W1f[(nt * 2 + 0) * 32 + lane];
    v16h bf1 = W1f[(nt * 2 + 1) * 32 + lane];
    v8f c = {};
    c = wmma_f16(a0, bf0, c);
    c = wmma_f16(a1, bf1, c);
    float bv = bias1[nt * 16 + n];
    int colD = nt * 16 + n;
    int ktD = colD >> 5;
#pragma unroll
    for (int r = 0; r < 8; ++r) {
      int rowD = r + 8 * half;
      hF[ktD * 512 + frag_pos_of(colD & 31, rowD)] = (_Float16)fmaxf(c[r] + bv, 0.f);
    }
  }
  __syncthreads();

  const v16h* hFv = (const v16h*)hF;
  v16h h0 = hFv[lane];
  v16h h1 = hFv[32 + lane];
  const v16h* W2f = (const v16h*)W2p;   // KT=2, NT=wn/16
  int wnt = wn >> 4;
  for (int nt = 0; nt < wnt; ++nt) {
    v16h bf0 = W2f[(nt * 2 + 0) * 32 + lane];
    v16h bf1 = W2f[(nt * 2 + 1) * 32 + lane];
    v8f c = {};
    c = wmma_f16(h0, bf0, c);
    c = wmma_f16(h1, bf1, c);
    float bv = bias2[nt * 16 + n];
#pragma unroll
    for (int r = 0; r < 8; ++r)
      wL[(r + 8 * half) * wn + nt * 16 + n] = c[r] + bv;
  }
  __syncthreads();

  if (lane < 16) {
    int e = tile * 16 + lane;
    if (e < E)
      tp_scatter(layer, node, ei[e], ei[E + e], &sh[e * 9], &wL[lane * wn], agg, cnt);
  }
}

// ---------------- pairwise MDN head ----------------

__global__ void pair_pre(const float* nodeL, const float* nodeR,
                         const float* W1 /*[32][16]*/, const float* b1,
                         float* Alig, float* Brec) {
  int i = blockIdx.x * blockDim.x + threadIdx.x;
  if (i < NLIG * 16) {
    int nn = i / 16, c = i % 16;
    float acc = b1[c];
    for (int k = 0; k < 16; ++k) acc += nodeL[nn * 48 + k] * W1[k * 16 + c];
    Alig[i] = acc;
  } else if (i < (NLIG + NREC) * 16) {
    int j = i - NLIG * 16;
    int nn = j / 16, c = j % 16;
    float acc = 0.f;
    for (int k = 0; k < 16; ++k) acc += nodeR[nn * 48 + k] * W1[(16 + k) * 16 + c];
    Brec[j] = acc;
  }
}

__global__ void __launch_bounds__(256)
pair_kernel(const float* Alig, const float* Brec,
            const _Float16* W2p /*frag KT=1,NT=1*/, const float* b2,
            const _Float16* Wheadp /*frag KT=1,NT=2*/,
            const float* pi_b, const float* sg_b, const float* mu_b,
            const float* lig_pos, const float* rec_pos, float* out) {
  __shared__ __align__(32) _Float16 s_h1[8][512];
  __shared__ __align__(32) _Float16 s_h2[8][512];
  __shared__ float s_o[8][16 * 32];
  int wave = threadIdx.x >> 5, lane = threadIdx.x & 31;
  int n = lane & 15, half = lane >> 4;
  int tileG = blockIdx.x * 8 + wave;
  int jt = tileG & 63;
  int iL = (tileG >> 6) & 127;
  int b  = tileG >> 13;
  int li = b * NLI + iL;
  int j0 = b * NTG + jt * 16;

  _Float16* h1 = s_h1[wave];
  _Float16* h2 = s_h2[wave];
  float*    so = s_o[wave];

  for (int idx = lane; idx < 512; idx += 32) {
    int r = idx >> 5, c = idx & 31;
    float v = 0.f;
    if (c < 16) v = fmaxf(Alig[li * 16 + c] + Brec[(j0 + r) * 16 + c], 0.f);
    h1[frag_pos_of(c, r)] = (_Float16)v;
    h2[idx] = (_Float16)0.f;
  }
  __builtin_prefetch(&Brec[(j0 + 1024) * 16], 0, 1);
  __syncthreads();

  {
    v16h a = ((const v16h*)h1)[lane];
    v16h bf = ((const v16h*)W2p)[lane];
    v8f c = {};
    c = wmma_f16(a, bf, c);
    float bv = b2[n];
#pragma unroll
    for (int r = 0; r < 8; ++r) {
      int rowD = r + 8 * half;
      h2[frag_pos_of(n, rowD)] = (_Float16)(c[r] + bv);
    }
  }
  __syncthreads();

  v16h a2 = ((const v16h*)h2)[lane];
#pragma unroll
  for (int nt = 0; nt < 2; ++nt) {
    v16h bf = ((const v16h*)Wheadp)[nt * 32 + lane];
    v8f c = {};
    c = wmma_f16(a2, bf, c);
    int col = nt * 16 + n;
    float bv = (col < 10) ? pi_b[col] : (col < 20) ? sg_b[col - 10]
             : (col < 30) ? mu_b[col - 20] : 0.f;
#pragma unroll
    for (int r = 0; r < 8; ++r) so[(r + 8 * half) * 32 + col] = c[r] + bv;
  }
  __syncthreads();

  if (lane < 16) {
    int r = lane;
    int j = jt * 16 + r;
    size_t pair = (size_t)b * (NLI * NTG) + (size_t)iL * NTG + j;
    const float* row = &so[r * 32];
    float mx = row[0];
    for (int k = 1; k < 10; ++k) mx = fmaxf(mx, row[k]);
    float ssum = 0.f, ex[10];
    for (int k = 0; k < 10; ++k) { ex[k] = expf(row[k] - mx); ssum += ex[k]; }
    float inv = 1.f / ssum;
    for (int k = 0; k < 10; ++k) out[pair * 10 + k] = ex[k] * inv;
    for (int k = 0; k < 10; ++k) {
      float v = row[10 + k];
      out[(size_t)NPAIR * 10 + pair * 10 + k] = (v > 0.f ? v : expf(v) - 1.f) + 1.1f;
    }
    for (int k = 0; k < 10; ++k) {
      float v = row[20 + k];
      out[(size_t)NPAIR * 20 + pair * 10 + k] = (v > 0.f ? v : expf(v) - 1.f) + 1.0f;
    }
    float dx = lig_pos[li * 3 + 0] - rec_pos[(b * NTG + j) * 3 + 0];
    float dy = lig_pos[li * 3 + 1] - rec_pos[(b * NTG + j) * 3 + 1];
    float dz = lig_pos[li * 3 + 2] - rec_pos[(b * NTG + j) * 3 + 2];
    out[(size_t)NPAIR * 30 + pair] = sqrtf(dx * dx + dy * dy + dz * dz);
  }
}

// ---------------- host orchestration ----------------

static inline size_t align_up(size_t x, size_t a) { return (x + a - 1) & ~(a - 1); }

extern "C" void kernel_launch(void* const* d_in, const int* in_sizes, int n_in,
                              void* d_out, int out_size, void* d_ws, size_t ws_size,
                              hipStream_t stream) {
  (void)in_sizes; (void)n_in; (void)out_size; (void)ws_size;
  char* ws = (char*)d_ws;
  size_t off = 0;
  auto alloc = [&](size_t bytes) -> void* {
    void* p = ws + off;
    off = align_up(off + bytes, 256);
    return p;
  };

  float* nodeL0 = (float*)alloc((size_t)NLIG * 48 * 4);
  float* nodeL1 = (float*)alloc((size_t)NLIG * 48 * 4);
  float* nodeR0 = (float*)alloc((size_t)NREC * 48 * 4);
  float* nodeR1 = (float*)alloc((size_t)NREC * 48 * 4);
  float* leaL   = (float*)alloc((size_t)ELIG * 16 * 4);
  float* shL    = (float*)alloc((size_t)ELIG * 9 * 4);
  float* reaR   = (float*)alloc((size_t)EREC * 16 * 4);
  float* shR    = (float*)alloc((size_t)EREC * 9 * 4);
  float* aggL   = (float*)alloc((size_t)NLIG * 48 * 4);
  float* cntL   = (float*)alloc((size_t)NLIG * 4);
  float* aggR   = (float*)alloc((size_t)NREC * 48 * 4);
  float* cntR   = (float*)alloc((size_t)NREC * 4);
  const int WN[2] = {320, 432};
  _Float16* W1p[4];
  _Float16* W2p[4];
  for (int i = 0; i < 4; ++i) W1p[i] = (_Float16*)alloc((size_t)2 * 3 * 512 * 2);
  for (int i = 0; i < 4; ++i) W2p[i] = (_Float16*)alloc((size_t)2 * (WN[i & 1] / 16) * 512 * 2);
  _Float16* mlpW2p = (_Float16*)alloc(512 * 2);
  _Float16* Wheadp = (_Float16*)alloc(2 * 512 * 2);
  float* Wcat = (float*)alloc(512 * 4);
  float* Alig = (float*)alloc((size_t)NLIG * 16 * 4);
  float* Brec = (float*)alloc((size_t)NREC * 16 * 4);

  const float* lig_node = (const float*)d_in[0];
  const float* rec_node = (const float*)d_in[1];
  const float* lig_pos  = (const float*)d_in[2];
  const float* rec_pos  = (const float*)d_in[3];
  const float* lig_bond = (const float*)d_in[4];
  const int*   lig_ei   = (const int*)d_in[5];
  const int*   rec_ei   = (const int*)d_in[6];

  auto B = [](long n) { return (unsigned)((n + 255) / 256); };

  // pack weight panels into fragment order (writes every element, no pre-zero)
  const int fcW1[4] = {15, 19, 23, 27};
  for (int i = 0; i < 4; ++i) {
    pack_b_frags<<<B(2 * 3 * 512), 256, 0, stream>>>(
        (const float*)d_in[fcW1[i]], 48, 48, 2, 3, W1p[i]);
    pack_b_frags<<<B((long)2 * (WN[i & 1] / 16) * 512), 256, 0, stream>>>(
        (const float*)d_in[fcW1[i] + 2], 48, WN[i & 1], 2, WN[i & 1] / 16, W2p[i]);
  }
  pack_b_frags<<<B(512), 256, 0, stream>>>((const float*)d_in[33], 16, 16, 1, 1, mlpW2p);
  build_whead<<<B(512), 256, 0, stream>>>((const float*)d_in[35], (const float*)d_in[37],
                                          (const float*)d_in[39], Wcat);
  pack_b_frags<<<B(2 * 512), 256, 0, stream>>>(Wcat, 16, 32, 1, 2, Wheadp);

  // edge features
  edge_feat_kernel<<<B(ELIG), 256, 0, stream>>>(lig_pos, lig_ei, lig_bond, 4, 5.0f,
      (const float*)d_in[7], (const float*)d_in[8], (const float*)d_in[9], (const float*)d_in[10],
      leaL, shL, ELIG);
  edge_feat_kernel<<<B(EREC), 256, 0, stream>>>(rec_pos, rec_ei, nullptr, 0, 30.0f,
      (const float*)d_in[11], (const float*)d_in[12], (const float*)d_in[13], (const float*)d_in[14],
      reaR, shR, EREC);

  node_init<<<B(NLIG * 48), 256, 0, stream>>>(lig_node, nodeL0, NLIG, 16);
  node_init<<<B(NREC * 48), 256, 0, stream>>>(rec_node, nodeR0, NREC, 16);

  struct G { const int* ei; int E; int N; const float* ea; const float* sh;
             float* n0; float* n1; float* agg; float* cnt; int wbase; };
  G gs[2] = {
    { lig_ei, ELIG, NLIG, leaL, shL, nodeL0, nodeL1, aggL, cntL, 0 },
    { rec_ei, EREC, NREC, reaR, shR, nodeR0, nodeR1, aggR, cntR, 2 } };
  for (int g = 0; g < 2; ++g) {
    float* cur = gs[g].n0;
    float* nxt = gs[g].n1;
    for (int l = 0; l < 2; ++l) {
      fill_f32<<<B((long)gs[g].N * 48), 256, 0, stream>>>(gs[g].agg, (long)gs[g].N * 48);
      fill_f32<<<B(gs[g].N), 256, 0, stream>>>(gs[g].cnt, gs[g].N);
      int wi = gs[g].wbase + l;
      conv_fused<<<(gs[g].E + 15) / 16, 32, 0, stream>>>(
          cur, gs[g].ei, gs[g].E, gs[g].ea, gs[g].sh,
          W1p[wi], (const float*)d_in[fcW1[wi] + 1],
          W2p[wi], (const float*)d_in[fcW1[wi] + 3],
          WN[l], l, gs[g].agg, gs[g].cnt);
      node_update<<<B((long)gs[g].N * 48), 256, 0, stream>>>(cur, gs[g].agg, gs[g].cnt, nxt, gs[g].N);
      float* t = cur; cur = nxt; nxt = t;
    }
  }
  // final nodes back in nodeL0 / nodeR0 after two swaps

  pair_pre<<<B((NLIG + NREC) * 16), 256, 0, stream>>>(nodeL0, nodeR0,
      (const float*)d_in[31], (const float*)d_in[32], Alig, Brec);

  pair_kernel<<<(NPAIR / 16) / 8, 256, 0, stream>>>(Alig, Brec,
      mlpW2p, (const float*)d_in[34], Wheadp,
      (const float*)d_in[36], (const float*)d_in[38], (const float*)d_in[40],
      lig_pos, rec_pos, (float*)d_out);
}
